// FullAttentionLayer_65833258713690
// MI455X (gfx1250) — matrix operand order repaired and checked by reading, not compiled, not run on hardware
//
#include <hip/hip_runtime.h>
#include <hip/hip_bf16.h>

typedef __bf16 bf16;
typedef __attribute__((ext_vector_type(16))) __bf16 v16bf;
typedef __attribute__((ext_vector_type(8)))  __bf16 v8bf;
typedef __attribute__((ext_vector_type(8)))  float  v8f;

#define BM 128
#define BN 256
#define BK 32
#define LSTR 40   // padded LDS row stride (bf16): 80B keeps 16B alignment, spreads banks

union Frag { v16bf v; v8bf h[2]; };

// Low 32 bits of a generic pointer into __shared__ == LDS byte offset
// (LDS aperture: ADDR[63:32] selects the aperture, ADDR[31:0] is the LDS addr).
__device__ inline unsigned lds_addr32(const void* p) {
    return (unsigned)(unsigned long long)p;
}

// gfx1250 async copy global->LDS, 16B per lane, tracked by ASYNCcnt.
__device__ inline void async_copy16(unsigned lds_dst, const void* gsrc) {
    asm volatile("global_load_async_to_lds_b128 %0, %1, off"
                 :: "v"(lds_dst), "v"((unsigned long long)gsrc)
                 : "memory");
}

__device__ inline void wait_async0() {
#if __has_builtin(__builtin_amdgcn_s_wait_asynccnt)
    __builtin_amdgcn_s_wait_asynccnt(0);
#else
    asm volatile("s_wait_asynccnt 0x0" ::: "memory");
#endif
}

// ---------------------------------------------------------------------------
// C[M,N](f32 and/or bf16) = A[M,K]bf16 row-major  x  B bf16 (+bias)(+relu)
//   bt==0: B is [K,N] row-major (transposed into LDS on fill, VGPR path)
//   bt==1: B is [N,K] row-major (A @ B^T; direct async fill)
// Block: 256 thr = 8 waves (2x4), block tile 128x256, wave tile 64x64,
// 16 v_wmma_f32_16x16x32_bf16 per K-step per wave.
// ---------------------------------------------------------------------------
__global__ __launch_bounds__(256)
void wmma_gemm_bf16(const bf16* __restrict__ A, const bf16* __restrict__ B,
                    float* __restrict__ Cf, bf16* __restrict__ Cb,
                    const float* __restrict__ bias,
                    int M, int N, int K, int bt, int relu)
{
    __shared__ bf16 As[2][BM * LSTR];   // 20.5 KB
    __shared__ bf16 Bs[2][BN * LSTR];   // 41 KB  (61.5 KB of the 320 KB WGP LDS)

    const int t    = threadIdx.x;
    const int lane = t & 31;
    const int wid  = t >> 5;            // 8 waves: 2 (M) x 4 (N)
    const int wm   = (wid >> 2) * 64;
    const int wn   = (wid & 3) * 64;
    const int lr   = lane & 15;
    const int lh   = lane >> 4;
    const int row0 = blockIdx.y * BM;
    const int col0 = blockIdx.x * BN;

    auto load_tiles = [&](int buf, int k0) {
        // A tile 128x32: async, 2 x 16B per thread
        #pragma unroll
        for (int i = 0; i < 2; ++i) {
            int r = (t >> 2) + i * 64;
            int c = (t & 3) * 8;
            async_copy16(lds_addr32(&As[buf][r * LSTR + c]),
                         A + (size_t)(row0 + r) * K + (k0 + c));
        }
        if (bt) {
            // B^T rows are K-contiguous: async, 4 x 16B per thread (256 rows)
            #pragma unroll
            for (int i = 0; i < 4; ++i) {
                int r = (t >> 2) + i * 64;
                int c = (t & 3) * 8;
                async_copy16(lds_addr32(&Bs[buf][r * LSTR + c]),
                             B + (size_t)(col0 + r) * K + (k0 + c));
            }
        } else {
            // B [K,N]: transpose into LDS so each column is K-contiguous
            #pragma unroll
            for (int i = 0; i < 4; ++i) {
                int kk = (t >> 5) + i * 8;
                int nn = (t & 31) * 8;
                v8bf v = *(const v8bf*)(B + (size_t)(k0 + kk) * N + (col0 + nn));
                #pragma unroll
                for (int e = 0; e < 8; ++e)
                    Bs[buf][(nn + e) * LSTR + kk] = v[e];
            }
        }
    };

    v8f acc[4][4] = {};

    load_tiles(0, 0);
    wait_async0();
    __syncthreads();

    const int nk = K / BK;
    for (int ki = 0; ki < nk; ++ki) {
        const int buf = ki & 1;
        if (ki + 1 < nk) load_tiles(buf ^ 1, (ki + 1) * BK);
        if (ki + 2 < nk) {  // gfx1250 global_prefetch_b8, two tiles ahead
            __builtin_prefetch(A + (size_t)(row0 + (t >> 2)) * K + (ki + 2) * BK + (t & 3) * 8, 0, 1);
            const bf16* bp = bt
                ? (B + (size_t)(col0 + (t >> 2)) * K + (ki + 2) * BK + (t & 3) * 8)
                : (B + (size_t)((ki + 2) * BK + (t >> 5)) * N + col0 + (t & 31) * 8);
            __builtin_prefetch(bp, 0, 1);
        }

        // Fragments per ISA 7.12.2 16-bit layouts
        Frag a[4], b[4];
        #pragma unroll
        for (int mt = 0; mt < 4; ++mt) {
            const bf16* p = &As[buf][(wm + mt * 16 + lr) * LSTR + lh * 8];
            a[mt].h[0] = *(const v8bf*)p;        // K = lh*8 .. +7
            a[mt].h[1] = *(const v8bf*)(p + 16); // K = 16+lh*8 .. +7
        }
        #pragma unroll
        for (int nt = 0; nt < 4; ++nt) {
            const bf16* p = &Bs[buf][(wn + nt * 16 + lr) * LSTR + lh * 8];
            b[nt].h[0] = *(const v8bf*)p;        // K = lh*8 .. +7
            b[nt].h[1] = *(const v8bf*)(p + 16); // K = 16+lh*8 .. +7
        }
        #pragma unroll
        for (int mt = 0; mt < 4; ++mt)
            #pragma unroll
            for (int nt = 0; nt < 4; ++nt)
                acc[mt][nt] = __builtin_amdgcn_wmma_f32_16x16x32_bf16(
                    false, a[mt].v, false, b[nt].v, (short)0, acc[mt][nt],
                    false, false);

        if (ki + 1 < nk) wait_async0();
        __syncthreads();
    }

    // Epilogue: VGPR r holds (M = lh*8 + r, N = lr) within each 16x16 tile
    #pragma unroll
    for (int nt = 0; nt < 4; ++nt) {
        int col = col0 + wn + nt * 16 + lr;
        float bv = bias ? bias[col] : 0.0f;
        #pragma unroll
        for (int mt = 0; mt < 4; ++mt) {
            int rb = row0 + wm + mt * 16 + lh * 8;
            #pragma unroll
            for (int r = 0; r < 8; ++r) {
                float x = acc[mt][nt][r] + bv;
                if (relu && x < 0.0f) x = 0.0f;
                size_t idx = (size_t)(rb + r) * N + col;
                if (Cf) Cf[idx] = x;
                if (Cb) Cb[idx] = (bf16)x;
            }
        }
    }
}

// ---------------------------------------------------------------------------
__device__ inline float block_sum(float v, float* red) {
    int t = threadIdx.x;
    __syncthreads();
    red[t] = v; __syncthreads();
    #pragma unroll
    for (int s = 128; s > 0; s >>= 1) {
        if (t < s) red[t] += red[t + s];
        __syncthreads();
    }
    return red[0];
}

__device__ inline float block_max(float v, float* red) {
    int t = threadIdx.x;
    __syncthreads();
    red[t] = v; __syncthreads();
    #pragma unroll
    for (int s = 128; s > 0; s >>= 1) {
        if (t < s) red[t] = fmaxf(red[t], red[t + s]);
        __syncthreads();
    }
    return red[0];
}

// Row softmax over n=4096 columns: 256 threads x 16 elems, fp32 in, bf16 out.
__global__ __launch_bounds__(256)
void softmax_rows(const float* __restrict__ S, bf16* __restrict__ P, int n)
{
    __shared__ float red[256];
    const int t = threadIdx.x;
    const size_t base = (size_t)blockIdx.x * n;

    float v[16];
    const float4* src = (const float4*)(S + base) + t * 4;
    #pragma unroll
    for (int i = 0; i < 4; ++i) {
        float4 x = src[i];
        v[i*4+0] = x.x; v[i*4+1] = x.y; v[i*4+2] = x.z; v[i*4+3] = x.w;
    }
    float m = -3.4e38f;
    #pragma unroll
    for (int i = 0; i < 16; ++i) m = fmaxf(m, v[i]);
    m = block_max(m, red);

    float s = 0.0f;
    #pragma unroll
    for (int i = 0; i < 16; ++i) { v[i] = __expf(v[i] - m); s += v[i]; }
    s = block_sum(s, red);
    float inv = 1.0f / s;

    bf16* dst = P + base + (size_t)t * 16;
    #pragma unroll
    for (int i = 0; i < 16; ++i) dst[i] = (bf16)(v[i] * inv);
}

// out = LayerNorm(X + R) with gamma/beta; d=1024: 256 threads x 4 elems.
__global__ __launch_bounds__(256)
void residual_layernorm(const float* __restrict__ X, const float* __restrict__ R,
                        const float* __restrict__ g, const float* __restrict__ be,
                        float* __restrict__ Of, bf16* __restrict__ Ob, int d)
{
    __shared__ float red[256];
    const int t = threadIdx.x;
    const size_t base = (size_t)blockIdx.x * d;

    float4 x = *(const float4*)(X + base + t * 4);
    float4 r = *(const float4*)(R + base + t * 4);
    float v[4] = { x.x + r.x, x.y + r.y, x.z + r.z, x.w + r.w };

    float s = v[0] + v[1] + v[2] + v[3];
    float mu = block_sum(s, red) / (float)d;

    float q = 0.0f;
    #pragma unroll
    for (int i = 0; i < 4; ++i) { float dv = v[i] - mu; q += dv * dv; }
    float var = block_sum(q, red) / (float)d;
    float inv = rsqrtf(var + 1e-5f);

    #pragma unroll
    for (int i = 0; i < 4; ++i) {
        int c = t * 4 + i;
        float o = (v[i] - mu) * inv * g[c] + be[c];
        Of[base + c] = o;
        if (Ob) Ob[base + c] = (bf16)o;
    }
}

__global__ __launch_bounds__(256)
void f32_to_bf16_kernel(const float* __restrict__ in, bf16* __restrict__ out, size_t n)
{
    size_t i = ((size_t)blockIdx.x * blockDim.x + threadIdx.x) * 4;
    if (i + 3 < n) {
        float4 v = *(const float4*)(in + i);
        out[i + 0] = (bf16)v.x; out[i + 1] = (bf16)v.y;
        out[i + 2] = (bf16)v.z; out[i + 3] = (bf16)v.w;
    }
}

// ---------------------------------------------------------------------------
extern "C" void kernel_launch(void* const* d_in, const int* in_sizes, int n_in,
                              void* d_out, int out_size, void* d_ws, size_t ws_size,
                              hipStream_t stream)
{
    const int N = 4096, D = 1024, H = 4096;

    const float* x   = (const float*)d_in[0];
    const float* Wq  = (const float*)d_in[1];
    const float* Wk  = (const float*)d_in[2];
    const float* Wv  = (const float*)d_in[3];
    const float* Wo  = (const float*)d_in[4];
    const float* W1  = (const float*)d_in[5];
    const float* b1  = (const float*)d_in[6];
    const float* W2  = (const float*)d_in[7];
    const float* b2  = (const float*)d_in[8];
    const float* g1  = (const float*)d_in[9];
    const float* be1 = (const float*)d_in[10];
    const float* g2  = (const float*)d_in[11];
    const float* be2 = (const float*)d_in[12];
    float* out = (float*)d_out;

    char* p = (char*)d_ws;
    auto alloc = [&](size_t bytes) -> void* {
        void* r = (void*)p;
        p += (bytes + 255) & ~(size_t)255;
        return r;
    };

    bf16* xb   = (bf16*)alloc((size_t)N * D * 2);
    bf16* Wqb  = (bf16*)alloc((size_t)D * H * 2);
    bf16* Wkb  = (bf16*)alloc((size_t)D * H * 2);
    bf16* Wvb  = (bf16*)alloc((size_t)D * H * 2);
    bf16* Wob  = (bf16*)alloc((size_t)H * D * 2);
    bf16* W1b  = (bf16*)alloc((size_t)D * H * 2);
    bf16* W2b  = (bf16*)alloc((size_t)H * D * 2);
    bf16* qb   = (bf16*)alloc((size_t)N * H * 2);
    bf16* kb   = (bf16*)alloc((size_t)N * H * 2);
    bf16* vb   = (bf16*)alloc((size_t)N * H * 2);
    float* Sf  = (float*)alloc((size_t)N * N * 4);
    bf16* Pb   = (bf16*)alloc((size_t)N * N * 2);
    bf16* AVb  = (bf16*)alloc((size_t)N * H * 2);
    float* attnf = (float*)alloc((size_t)N * D * 4);
    float* hf  = (float*)alloc((size_t)N * D * 4);
    bf16* hb   = (bf16*)alloc((size_t)N * D * 2);
    bf16* ff1b = (bf16*)alloc((size_t)N * H * 2);
    float* ff2f = (float*)alloc((size_t)N * D * 4);

    dim3 blk(256);
    auto conv = [&](const float* in, bf16* o, size_t n) {
        f32_to_bf16_kernel<<<dim3((unsigned)(n / 1024)), blk, 0, stream>>>(in, o, n);
    };
    auto gemm = [&](const bf16* A, const bf16* B, float* Cf, bf16* Cb,
                    const float* bias, int M, int Nn, int K, int bt, int relu) {
        dim3 grid(Nn / BN, M / BM);
        wmma_gemm_bf16<<<grid, blk, 0, stream>>>(A, B, Cf, Cb, bias, M, Nn, K, bt, relu);
    };

    // bf16 casts of activations + weights
    conv(x,  xb,  (size_t)N * D);
    conv(Wq, Wqb, (size_t)D * H);
    conv(Wk, Wkb, (size_t)D * H);
    conv(Wv, Wvb, (size_t)D * H);
    conv(Wo, Wob, (size_t)H * D);
    conv(W1, W1b, (size_t)D * H);
    conv(W2, W2b, (size_t)H * D);

    // q, k, v
    gemm(xb, Wqb, nullptr, qb, nullptr, N, H, D, 0, 0);
    gemm(xb, Wkb, nullptr, kb, nullptr, N, H, D, 0, 0);
    gemm(xb, Wvb, nullptr, vb, nullptr, N, H, D, 0, 0);

    // logits = q @ k^T   (k rows are K-contiguous -> bt/async path)
    gemm(qb, kb, Sf, nullptr, nullptr, N, N, H, 1, 0);

    // softmax rows -> bf16 probabilities
    softmax_rows<<<dim3(N), blk, 0, stream>>>(Sf, Pb, N);

    // attn = (P @ v) @ Wo
    gemm(Pb, vb, nullptr, AVb, nullptr, N, H, N, 0, 0);
    gemm(AVb, Wob, attnf, nullptr, nullptr, N, D, H, 0, 0);

    // h = LN(x + attn)
    residual_layernorm<<<dim3(N), blk, 0, stream>>>(x, attnf, g1, be1, hf, hb, D);

    // ff = relu(h @ W1 + b1) @ W2 + b2
    gemm(hb, W1b, nullptr, ff1b, b1, N, H, D, 0, 1);
    gemm(ff1b, W2b, ff2f, nullptr, b2, N, D, H, 0, 0);

    // out = LN(h + ff)
    residual_layernorm<<<dim3(N), blk, 0, stream>>>(hf, ff2f, g2, be2, out, nullptr, D);
}
